// MVT_Mamba_62603443307010
// MI455X (gfx1250) — matrix operand order repaired and checked by reading, not compiled, not run on hardware
//
#include <hip/hip_runtime.h>
#include <hip/hip_bf16.h>

typedef __bf16 bf16_t;
typedef __attribute__((ext_vector_type(16))) __bf16 v16bf;
typedef __attribute__((ext_vector_type(8)))  __bf16 v8bf;
typedef __attribute__((ext_vector_type(8)))  float  v8f;

constexpr int Bb  = 4;
constexpr int Ll  = 2048;
constexpr int DM  = 512;
constexpr int DS  = 16;
constexpr int DC  = 4;
constexpr int DI  = 1024;           // EXPAND * D_MODEL
constexpr int DTR = 32;             // dt_rank
constexpr int NT  = Bb * Ll;        // 8192 tokens
constexpr int XZW = 2 * DI;         // 2048
constexpr int XDW = DTR + 2 * DS;   // 64

__device__ __forceinline__ float silu_f(float x)     { return x / (1.f + __expf(-x)); }
__device__ __forceinline__ float softplus_f(float x) { return (x > 20.f) ? x : log1pf(__expf(x)); }

// A-fragment: lane holds row (lane&15); 8 bf16 at k + (lane>>4)*8 and 8 more at +16
__device__ __forceinline__ v16bf load_fragA(const bf16_t* p) {
  v8bf lo = *(const v8bf*)(p);
  v8bf hi = *(const v8bf*)(p + 16);
  return __builtin_shufflevector(lo, hi, 0,1,2,3,4,5,6,7,8,9,10,11,12,13,14,15);
}
// B-fragment: lane holds col (lane&15); 8 bf16 at k + (lane>>4)*8 and 8 more at +16
__device__ __forceinline__ v16bf load_fragB(const bf16_t* p) {
  v8bf lo = *(const v8bf*)(p);
  v8bf hi = *(const v8bf*)(p + 16);
  return __builtin_shufflevector(lo, hi, 0,1,2,3,4,5,6,7,8,9,10,11,12,13,14,15);
}
__device__ __forceinline__ v8f wmma_bf16(v16bf a, v16bf b, v8f c) {
  return __builtin_amdgcn_wmma_f32_16x16x32_bf16(false, a, false, b, (short)0, c, false, false);
}

// ---------------------------------------------------------------- fp32 -> bf16
__global__ void cvt_f32_to_bf16(const float* __restrict__ s, bf16_t* __restrict__ d, long long n) {
  long long i = (long long)blockIdx.x * blockDim.x + threadIdx.x;
  if (i < n) d[i] = (bf16_t)s[i];
}

// -------------------------------------------------- WMMA GEMM: C = A @ B^T
// A: [M,K] bf16 row-major (lda), B: [N,K] bf16 row-major (ldb), C: [M,N] f32 (ldc)
// grid = (M/256, N/64), block = 256 (8 waves); wave -> 32x64 tile, K step 32.
// Ping-pong double buffering (fragment sets X/Y, no register copies).
// act==1: C = softplus(C + bias[n])
__global__ void __launch_bounds__(256)
gemm_wmma_bf16(const bf16_t* __restrict__ A, int lda,
               const bf16_t* __restrict__ Bw, int ldb,
               float* __restrict__ C, int ldc,
               int K, const float* __restrict__ bias, int act) {
  const int lane = threadIdx.x & 31;
  const int wid  = threadIdx.x >> 5;
  const int lm   = lane & 15;
  const int lh   = lane >> 4;
  const int mtop  = (blockIdx.x * 8 + wid) * 32;
  const int nbase = blockIdx.y * 64;

  const bf16_t* arow0 = A + (size_t)(mtop + lm) * lda + lh * 8;
  const bf16_t* arow1 = arow0 + (size_t)16 * lda;
  const bf16_t* bbase = Bw + (size_t)(nbase + lm) * ldb + lh * 8;

  v8f acc[2][4] = {};
  v16bf xa0, xa1, xb[4];       // fragment set X
  v16bf ya0, ya1, yb[4];       // fragment set Y

  auto loadX = [&](int k) {
    xa0 = load_fragA(arow0 + k); xa1 = load_fragA(arow1 + k);
#pragma unroll
    for (int nt = 0; nt < 4; ++nt) xb[nt] = load_fragB(bbase + (size_t)nt * 16 * ldb + k);
  };
  auto loadY = [&](int k) {
    ya0 = load_fragA(arow0 + k); ya1 = load_fragA(arow1 + k);
#pragma unroll
    for (int nt = 0; nt < 4; ++nt) yb[nt] = load_fragB(bbase + (size_t)nt * 16 * ldb + k);
  };
  auto computeX = [&]() {
#pragma unroll
    for (int nt = 0; nt < 4; ++nt) {
      acc[0][nt] = wmma_bf16(xa0, xb[nt], acc[0][nt]);
      acc[1][nt] = wmma_bf16(xa1, xb[nt], acc[1][nt]);
    }
  };
  auto computeY = [&]() {
#pragma unroll
    for (int nt = 0; nt < 4; ++nt) {
      acc[0][nt] = wmma_bf16(ya0, yb[nt], acc[0][nt]);
      acc[1][nt] = wmma_bf16(ya1, yb[nt], acc[1][nt]);
    }
  };

  const int NK = K >> 5;       // number of 32-wide K steps (16, 32, or 1 here)
  int i = 0;
  loadX(0);
  for (; i + 2 <= NK - 1; i += 2) {   // at least two steps beyond i remain
    loadY((i + 1) * 32);
    computeX();                       // step i   (Y loads overlap these WMMAs)
    loadX((i + 2) * 32);
    computeY();                       // step i+1 (X loads overlap these WMMAs)
  }
  if (i == NK - 2) {                  // exactly one step pending after X
    loadY((i + 1) * 32);
    computeX();
    computeY();
  } else {                            // i == NK-1: X is the last step
    computeX();
  }

#pragma unroll
  for (int mt = 0; mt < 2; ++mt) {
#pragma unroll
    for (int nt = 0; nt < 4; ++nt) {
      const int col = nbase + nt * 16 + lm;
#pragma unroll
      for (int r = 0; r < 8; ++r) {
        const int row = mtop + mt * 16 + r + lh * 8;
        float v = acc[mt][nt][r];
        if (act == 1) v = softplus_f(v + bias[col]);
        C[(size_t)row * ldc + col] = v;
      }
    }
  }
}

// ----------------------------------- causal depthwise conv (K=4) + SiLU
// reads x half of xz (row stride XZW); rev=1 consumes the time-reversed sequence
__global__ void conv_silu_kernel(const float* __restrict__ xz,
                                 const float* __restrict__ w,
                                 const float* __restrict__ cb,
                                 float* __restrict__ xo, int rev) {
  long long idx = (long long)blockIdx.x * blockDim.x + threadIdx.x;
  if (idx >= (long long)NT * DI) return;
  const int d = (int)(idx & (DI - 1));
  const int l = (int)((idx >> 10) & (Ll - 1));
  const int b = (int)(idx >> 21);
  float acc = cb[d];
#pragma unroll
  for (int j = 0; j < DC; ++j) {
    int ls = l - (DC - 1) + j;
    if (ls >= 0) {
      int lsrc = rev ? (Ll - 1 - ls) : ls;
      acc += w[d * DC + j] * xz[((size_t)b * Ll + lsrc) * XZW + d];
    }
  }
  xo[((size_t)b * Ll + l) * DI + d] = silu_f(acc);
}

// -------------------------------------------------- selective scan
// one thread per (b, d): 16-state recurrence in registers over L;
// per-step B/C (shared by all d of a batch) staged in LDS, read as broadcasts.
// Next chunk's B/C rows are prefetched (global_prefetch) while this chunk runs.
__global__ void __launch_bounds__(256)
scan_kernel(const float* __restrict__ dt, const float* __restrict__ xc,
            const float* __restrict__ xdbl, const float* __restrict__ A_log,
            float* __restrict__ y) {
  __shared__ float sBC[64][32];          // [step][B(0..15) | C(16..31)]
  const int tid = threadIdx.x;
  const int d   = blockIdx.x * 256 + tid;
  const int b   = blockIdx.y;

  float Ar[DS], hs[DS];
#pragma unroll
  for (int n = 0; n < DS; ++n) { Ar[n] = -__expf(A_log[d * DS + n]); hs[n] = 0.f; }

  for (int l0 = 0; l0 < Ll; l0 += 64) {
    __syncthreads();                     // WAR guard on sBC between chunks
    for (int i = tid; i < 64 * 32; i += 256) {
      int s = i >> 5, c = i & 31;
      sBC[s][c] = xdbl[((size_t)b * Ll + l0 + s) * XDW + DTR + c];
    }
    if (l0 + 64 < Ll && tid < 64) {      // pull next chunk's B/C toward the WGP
      __builtin_prefetch(xdbl + ((size_t)b * Ll + l0 + 64 + tid) * XDW + DTR, 0, 1);
    }
    __syncthreads();
    for (int s = 0; s < 64; ++s) {
      const size_t r = ((size_t)b * Ll + l0 + s) * DI + d;
      const float dt_t = dt[r];
      const float x_t  = xc[r];
      const float dtx  = dt_t * x_t;
      float yv = 0.f;
#pragma unroll
      for (int n = 0; n < DS; ++n) {
        float dA = __expf(dt_t * Ar[n]);
        hs[n] = dA * hs[n] + dtx * sBC[s][n];
        yv += hs[n] * sBC[s][16 + n];
      }
      y[r] = yv;
    }
  }
}

// -------------------- gate + skip + direction merge, emit bf16 for out-proj
__global__ void combine_kernel(const float* __restrict__ xz,
                               const float* __restrict__ yf, const float* __restrict__ yr,
                               const float* __restrict__ xf, const float* __restrict__ xr,
                               const float* __restrict__ Df, const float* __restrict__ Dr,
                               bf16_t* __restrict__ out) {
  long long idx = (long long)blockIdx.x * blockDim.x + threadIdx.x;
  if (idx >= (long long)NT * DI) return;
  const int d = (int)(idx & (DI - 1));
  const int l = (int)((idx >> 10) & (Ll - 1));
  const int b = (int)(idx >> 21);
  const size_t rf = ((size_t)b * Ll + l) * DI + d;
  const size_t rr = ((size_t)b * Ll + (Ll - 1 - l)) * DI + d;
  const float zv = xz[((size_t)b * Ll + l) * XZW + DI + d];
  const float g  = silu_f(zv);
  const float v  = (yf[rf] + Df[d] * xf[rf] + yr[rr] + Dr[d] * xr[rr]) * g;
  out[rf] = (bf16_t)v;
}

// ============================================================ host side
extern "C" void kernel_launch(void* const* d_in, const int* in_sizes, int n_in,
                              void* d_out, int out_size, void* d_ws, size_t ws_size,
                              hipStream_t stream) {
  const float* hid    = (const float*)d_in[0];
  const float* W_in   = (const float*)d_in[1];
  const float* W_out  = (const float*)d_in[2];
  const float* conv_w[2] = {(const float*)d_in[3],  (const float*)d_in[10]};
  const float* conv_b[2] = {(const float*)d_in[4],  (const float*)d_in[11]};
  const float* W_x[2]    = {(const float*)d_in[5],  (const float*)d_in[12]};
  const float* W_dt[2]   = {(const float*)d_in[6],  (const float*)d_in[13]};
  const float* b_dt[2]   = {(const float*)d_in[7],  (const float*)d_in[14]};
  const float* A_log[2]  = {(const float*)d_in[8],  (const float*)d_in[15]};
  const float* Dp[2]     = {(const float*)d_in[9],  (const float*)d_in[16]};
  float* out = (float*)d_out;
  (void)in_sizes; (void)n_in; (void)out_size; (void)ws_size;

  // bump-allocate workspace (~267 MB), 256B aligned
  char* p = (char*)d_ws;
  auto alloc = [&](size_t bytes) { void* r = (void*)p; p += (bytes + 255) & ~(size_t)255; return r; };
  bf16_t* h_bf       = (bf16_t*)alloc((size_t)NT * DM * 2);
  bf16_t* Win_bf     = (bf16_t*)alloc((size_t)XZW * DM * 2);
  bf16_t* Wout_bf    = (bf16_t*)alloc((size_t)DM * DI * 2);
  bf16_t* Wx_bf[2]   = {(bf16_t*)alloc((size_t)XDW * DI * 2), (bf16_t*)alloc((size_t)XDW * DI * 2)};
  bf16_t* Wdt_bf[2]  = {(bf16_t*)alloc((size_t)DI * DTR * 2), (bf16_t*)alloc((size_t)DI * DTR * 2)};
  float*  xz         = (float*)alloc((size_t)NT * XZW * 4);
  float*  xconv[2]   = {(float*)alloc((size_t)NT * DI * 4), (float*)alloc((size_t)NT * DI * 4)};
  float*  ybuf[2]    = {(float*)alloc((size_t)NT * DI * 4), (float*)alloc((size_t)NT * DI * 4)};
  bf16_t* xconvbf_sh = (bf16_t*)alloc((size_t)NT * DI * 2);   // per-dir scratch; also ycomb_bf
  float*  xdbl_sh    = (float*)alloc((size_t)NT * XDW * 4);
  bf16_t* xdblbf_sh  = (bf16_t*)alloc((size_t)NT * XDW * 2);
  float*  dt_sh      = (float*)alloc((size_t)NT * DI * 4);
  bf16_t* ycomb_bf   = xconvbf_sh;  // free after last GEMM2, reused for out-proj input

  auto cvt = [&](const float* s, bf16_t* d, long long n) {
    cvt_f32_to_bf16<<<(unsigned)((n + 255) / 256), 256, 0, stream>>>(s, d, n);
  };

  // precision downconvert for tensor-core path
  cvt(hid,   h_bf,    (long long)NT * DM);
  cvt(W_in,  Win_bf,  (long long)XZW * DM);
  cvt(W_out, Wout_bf, (long long)DM * DI);
  for (int dir = 0; dir < 2; ++dir) {
    cvt(W_x[dir],  Wx_bf[dir],  (long long)XDW * DI);
    cvt(W_dt[dir], Wdt_bf[dir], (long long)DI * DTR);
  }

  const dim3 gblk(256);
  const unsigned nElemBlocks = (unsigned)(((long long)NT * DI) / 256);

  // shared input projection: xz[8192,2048] = h @ W_in^T   (K=512)
  gemm_wmma_bf16<<<dim3(NT / 256, XZW / 64), gblk, 0, stream>>>(
      h_bf, DM, Win_bf, DM, xz, XZW, DM, nullptr, 0);

  for (int dir = 0; dir < 2; ++dir) {
    // depthwise causal conv + SiLU (rev consumes time-reversed x)
    conv_silu_kernel<<<nElemBlocks, 256, 0, stream>>>(xz, conv_w[dir], conv_b[dir], xconv[dir], dir);
    cvt(xconv[dir], xconvbf_sh, (long long)NT * DI);
    // x_dbl[8192,64] = x @ W_x^T  (K=1024)
    gemm_wmma_bf16<<<dim3(NT / 256, XDW / 64), gblk, 0, stream>>>(
        xconvbf_sh, DI, Wx_bf[dir], DI, xdbl_sh, XDW, DI, nullptr, 0);
    cvt(xdbl_sh, xdblbf_sh, (long long)NT * XDW);
    // dt[8192,1024] = softplus(dt_r @ W_dt^T + b_dt)  (K=32, dt_r = first 32 cols of x_dbl)
    gemm_wmma_bf16<<<dim3(NT / 256, DI / 64), gblk, 0, stream>>>(
        xdblbf_sh, XDW, Wdt_bf[dir], DTR, dt_sh, DI, DTR, b_dt[dir], 1);
    // selective scan
    scan_kernel<<<dim3(DI / 256, Bb), 256, 0, stream>>>(
        dt_sh, xconv[dir], xdbl_sh, A_log[dir], ybuf[dir]);
  }

  // gate, skip, merge directions -> bf16
  combine_kernel<<<nElemBlocks, 256, 0, stream>>>(
      xz, ybuf[0], ybuf[1], xconv[0], xconv[1], Dp[0], Dp[1], ycomb_bf);

  // out[8192,512] = y_comb @ W_out^T  (K=1024)
  gemm_wmma_bf16<<<dim3(NT / 256, DM / 64), gblk, 0, stream>>>(
      ycomb_bf, DI, Wout_bf, DI, out, DM, DI, nullptr, 0);
}
